// ImplicitLatentAutoEncoder_29128468202040
// MI455X (gfx1250) — matrix operand order repaired and checked by reading, not compiled, not run on hardware
//
#include <hip/hip_runtime.h>
#include <cstdint>
#include <cstddef>

// ---------------------------------------------------------------------------
// ImplicitLatentAutoEncoder forward for MI455X (gfx1250, wave32, WMMA).
// All GEMMs: f16 inputs (A from activations, B from pre-transposed weights),
// f32 accumulation via v_wmma_f32_16x16x32_f16, fused bias/LayerNorm/ReLU.
// ---------------------------------------------------------------------------

typedef _Float16 h8   __attribute__((ext_vector_type(8)));
typedef _Float16 v16h __attribute__((ext_vector_type(16)));
typedef float    v8f  __attribute__((ext_vector_type(8)));

static __device__ __forceinline__ v16h cat8(h8 lo, h8 hi) {
  return __builtin_shufflevector(lo, hi, 0, 1, 2, 3, 4, 5, 6, 7,
                                 8, 9, 10, 11, 12, 13, 14, 15);
}

// Per-lane A-row segment base pointers.
// amode 0: plain row-major [M,K]
// amode 1: edge-pair gather: row -> (b,p); segments = node-latents of src / nbr
// amode 2: concat of two row-major buffers (widths segK, seg2K)
static __device__ __forceinline__ void a_row_ptrs(
    const _Float16* __restrict__ A, const _Float16* __restrict__ A2,
    int amode, int segK, int seg2K, int rowOff, int r, int K,
    const _Float16** p0, const _Float16** p1) {
  if (amode == 0) {
    *p0 = A + (size_t)r * K;
    *p1 = *p0;
  } else if (amode == 1) {
    int rg = rowOff + r;
    int b = rg / 240, p = rg % 240;     // P = O*(O-1) = 240 pairs, src-major
    int s = p / 15,   q = p % 15;
    int nb = q + (q >= s ? 1 : 0);      // neighbor index skipping diagonal
    *p0 = A + (size_t)(b * 16 + s)  * segK;
    *p1 = A + (size_t)(b * 16 + nb) * segK;
  } else {
    *p0 = A  + (size_t)r * segK;
    *p1 = A2 + (size_t)r * seg2K;
  }
}

// One 16x16 output tile: D += A(16xK) * B(Kx16), K-chunks of 32, f16 WMMA.
// CDNA5 A layout: lane l holds row l&15; k-halves chosen by l>>4 (two 8xf16
// contiguous runs per 32-chunk). B layout: lane l holds col l&15, same two
// 8xf16 runs as A (k-halves chosen by l>>4). Chunk offsets are multiples of 8
// f16 (16B) and never straddle a concat-segment boundary (all seg widths and
// K are multiples of 32).
static __device__ __forceinline__ v8f wmma_rowtile(
    const _Float16* __restrict__ p0, const _Float16* __restrict__ p1, int segK,
    const _Float16* __restrict__ bp, int K, int khalf, int kb) {
  v8f acc = {0.f, 0.f, 0.f, 0.f, 0.f, 0.f, 0.f, 0.f};
  for (int kc = 0; kc < K; kc += 32) {
    const int ka0 = kc + khalf;
    const int ka1 = kc + 16 + khalf;
    const _Float16* a0 = (ka0 < segK) ? (p0 + ka0) : (p1 + (ka0 - segK));
    const _Float16* a1 = (ka1 < segK) ? (p0 + ka1) : (p1 + (ka1 - segK));
    h8 alo = *(const h8*)a0;
    h8 ahi = *(const h8*)a1;
    h8 blo = *(const h8*)(bp + kc + khalf);
    h8 bhi = *(const h8*)(bp + kc + 16 + khalf);
    acc = __builtin_amdgcn_wmma_f32_16x16x32_f16(
        /*neg_a=*/false, cat8(alo, ahi), /*neg_b=*/false, cat8(blo, bhi),
        /*c_mod=*/(short)0, acc, /*reuse_a=*/false, /*reuse_b=*/false);
  }
  return acc;
}

// GEMM + bias + LayerNorm + ReLU, f16 output. One block = 16 rows x full N.
// blockDim = 2*N (N/16 waves), dynamic LDS = 16*N floats.
__global__ __launch_bounds__(512) void k_gemm_ln(
    const _Float16* __restrict__ A, const _Float16* __restrict__ A2,
    int amode, int segK, int seg2K, int rowOff,
    const _Float16* __restrict__ Wt, const float* __restrict__ bias,
    const float* __restrict__ gamma, const float* __restrict__ beta,
    _Float16* __restrict__ Y, int N, int K) {
  extern __shared__ float ylds[];                  // 16 x N tile
  __shared__ float sred[16 * 32];
  __shared__ float sred2[16 * 32];
  __shared__ float srow[32];
  const int tid  = threadIdx.x;
  const int lane = tid & 31;
  const int wave = tid >> 5;
  const int m0 = blockIdx.x * 16;
  const int n0 = wave * 16;
  const int khalf = (lane >> 4) << 3;
  const int kb    = (lane >> 4) << 4;
  const _Float16 *p0, *p1;
  a_row_ptrs(A, A2, amode, segK, seg2K, rowOff, m0 + (lane & 15), K, &p0, &p1);
  const _Float16* bp = Wt + (size_t)(n0 + (lane & 15)) * K;
  const int effSeg = (amode == 0) ? K : segK;
  v8f acc = wmma_rowtile(p0, p1, effSeg, bp, K, khalf, kb);

  // Stage D (+bias) to LDS. D layout: lane holds col lane&15, rows 8*(l>>4)+v.
  const int dn = lane & 15;
  const int mb = (lane >> 4) << 3;
  const float bv = bias[n0 + dn];
#pragma unroll
  for (int v = 0; v < 8; ++v)
    ylds[(mb + v) * N + n0 + dn] = acc[v] + bv;
  __syncthreads();

  // LayerNorm over full row (feature dim N). tpr = N/8 threads per row.
  const int tpr = blockDim.x >> 4;
  const int c  = tid % tpr;
  const int rr = tid / tpr;
  float s1 = 0.f, s2 = 0.f;
  for (int n = c; n < N; n += tpr) {
    float v = ylds[rr * N + n];
    s1 += v;
    s2 += v * v;
  }
  sred[rr * 32 + c]  = s1;
  sred2[rr * 32 + c] = s2;
  __syncthreads();
  if (c == 0) {
    float a1 = 0.f, a2 = 0.f;
    for (int j = 0; j < tpr; ++j) { a1 += sred[rr * 32 + j]; a2 += sred2[rr * 32 + j]; }
    float mean = a1 / (float)N;
    float var  = a2 / (float)N - mean * mean;
    srow[rr]      = mean;
    srow[16 + rr] = rsqrtf(var + 1e-5f);
  }
  __syncthreads();
  const float mean = srow[rr];
  const float rstd = srow[16 + rr];
  for (int n = c; n < N; n += tpr) {
    float v = (ylds[rr * N + n] - mean) * rstd * gamma[n] + beta[n];
    v = v > 0.f ? v : 0.f;
    Y[(size_t)(m0 + rr) * N + n] = (_Float16)v;
  }
}

// GEMM + bias, no LN. One wave per 16x16 tile.
// outmode 0: f16 row-major [M,N]; outmode 1: f32 with (B,O,T,E)->(B,T,O,E)
// transpose fused into the store (final layer).
__global__ __launch_bounds__(32) void k_gemm(
    const _Float16* __restrict__ A, const _Float16* __restrict__ A2,
    int amode, int segK, int seg2K, int rowOff,
    const _Float16* __restrict__ Wt, const float* __restrict__ bias,
    void* __restrict__ Yout, int N, int K, int outmode) {
  const int lane = threadIdx.x & 31;
  const int m0 = blockIdx.x * 16;
  const int n0 = blockIdx.y * 16;
  const int khalf = (lane >> 4) << 3;
  const int kb    = (lane >> 4) << 4;
  const _Float16 *p0, *p1;
  a_row_ptrs(A, A2, amode, segK, seg2K, rowOff, m0 + (lane & 15), K, &p0, &p1);
  const _Float16* bp = Wt + (size_t)(n0 + (lane & 15)) * K;
  const int effSeg = (amode == 0) ? K : segK;
  v8f acc = wmma_rowtile(p0, p1, effSeg, bp, K, khalf, kb);
  const int dn = lane & 15;
  const int mb = (lane >> 4) << 3;
  const float bv = bias[n0 + dn];
  if (outmode == 0) {
    _Float16* Y = (_Float16*)Yout;
#pragma unroll
    for (int v = 0; v < 8; ++v)
      Y[(size_t)(m0 + mb + v) * N + n0 + dn] = (_Float16)(acc[v] + bv);
  } else {
    float* Y = (float*)Yout;
#pragma unroll
    for (int v = 0; v < 8; ++v) {
      int m = m0 + mb + v, n = n0 + dn;
      int b = m >> 4, o = m & 15, t = n >> 4, e = n & 15;
      Y[(size_t)(((b * 32 + t) * 16 + o) * 16 + e)] = acc[v] + bv;
    }
  }
}

// x (B,T,O,E) f32 -> nodes (B*O, 512) f16
__global__ void k_pack_nodes(const float* __restrict__ x,
                             _Float16* __restrict__ nodes) {
  int i = blockIdx.x * blockDim.x + threadIdx.x;   // 0..2M-1
  int b = i >> 13, t = (i >> 8) & 31, o = (i >> 4) & 15, e = i & 15;
  nodes[(size_t)b * 8192 + o * 512 + t * 16 + e] = (_Float16)x[i];
}

// W (din,dout) f32 -> Wt (dout,din) f16 (column-of-W becomes contiguous row)
__global__ void k_prep_weight(const float* __restrict__ W,
                              _Float16* __restrict__ Wt, int din, int dout,
                              int cnt) {
  int i = blockIdx.x * blockDim.x + threadIdx.x;
  if (i >= cnt) return;
  int k = i / dout, n = i % dout;
  Wt[(size_t)n * din + k] = (_Float16)W[i];
}

// e rows (b,p) b-major, p src-major, width 256 -> per-src mean over 15 neighbors, f16.
// One chunk = 64 batches; i over 64*16*256.
__global__ void k_edge_mean(const _Float16* __restrict__ Ein,
                            _Float16* __restrict__ Em, int bOff) {
  int i = blockIdx.x * blockDim.x + threadIdx.x;
  int h = i & 255, so = (i >> 8) & 15, b = i >> 12;
  const _Float16* base = Ein + (size_t)(b * 240 + so * 15) * 256 + h;
  float s = 0.f;
#pragma unroll
  for (int q = 0; q < 15; ++q) s += (float)base[(size_t)q * 256];
  Em[(size_t)((bOff + b) * 16 + so) * 256 + h] = (_Float16)(s * (1.f / 15.f));
}

extern "C" void kernel_launch(void* const* d_in, const int* in_sizes, int n_in,
                              void* d_out, int out_size, void* d_ws,
                              size_t ws_size, hipStream_t stream) {
  (void)in_sizes; (void)n_in; (void)out_size; (void)ws_size;
  char* ws = (char*)d_ws;
  size_t off = 0;
  auto alloc = [&](size_t bytes) {
    size_t r = off;
    off += (bytes + 255) & ~(size_t)255;
    return r;
  };

  const size_t nodes_off = alloc((size_t)4096 * 512 * 2);
  struct WI { int idx, din, dout; };
  const WI wi[15] = {
      {1, 1024, 256}, {5, 256, 256},  {9, 256, 256},                  // ee
      {11, 768, 256}, {15, 256, 128}, {19, 128, 64}, {23, 64, 32},    // ne
      {25, 64, 32},   {29, 32, 64},   {33, 64, 256}, {37, 256, 256},  // ed
      {39, 288, 256}, {43, 256, 128}, {47, 128, 64}, {51, 64, 512}};  // nd
  size_t wt_off[15];
  for (int i = 0; i < 15; ++i)
    wt_off[i] = alloc((size_t)wi[i].din * wi[i].dout * 2);
  const size_t h1_off  = alloc((size_t)15360 * 256 * 2);  // edge ping
  const size_t h2_off  = alloc((size_t)15360 * 256 * 2);  // edge pong
  const size_t em_off  = alloc((size_t)4096 * 256 * 2);   // encoder edge mean
  const size_t emd_off = alloc((size_t)4096 * 256 * 2);   // decoder edge mean
  const size_t g1_off  = alloc((size_t)4096 * 256 * 2);   // node ping
  const size_t g2_off  = alloc((size_t)4096 * 128 * 2);   // node pong
  const size_t z_off   = alloc((size_t)4096 * 32 * 2);    // latent z

  _Float16* nodes = (_Float16*)(ws + nodes_off);
  _Float16* h1  = (_Float16*)(ws + h1_off);
  _Float16* h2  = (_Float16*)(ws + h2_off);
  _Float16* em  = (_Float16*)(ws + em_off);
  _Float16* emd = (_Float16*)(ws + emd_off);
  _Float16* g1  = (_Float16*)(ws + g1_off);
  _Float16* g2  = (_Float16*)(ws + g2_off);
  _Float16* zb  = (_Float16*)(ws + z_off);

  auto Wp = [&](int i) { return (const _Float16*)(ws + wt_off[i]); };
  auto Fp = [&](int i) { return (const float*)d_in[i]; };

  // ---- prep: pack x into f16 node rows; transpose+convert all weights ----
  k_pack_nodes<<<8192, 256, 0, stream>>>((const float*)d_in[0], nodes);
  for (int i = 0; i < 15; ++i) {
    int cnt = wi[i].din * wi[i].dout;
    k_prep_weight<<<(cnt + 255) / 256, 256, 0, stream>>>(
        Fp(wi[i].idx), (_Float16*)(ws + wt_off[i]), wi[i].din, wi[i].dout, cnt);
  }

  auto gemm_ln = [&](const _Float16* A, const _Float16* A2, int amode, int segK,
                     int seg2K, int rowOff, int wt, int bidx, _Float16* Y,
                     int M, int N, int K) {
    k_gemm_ln<<<dim3(M / 16), dim3(2 * N), (size_t)16 * N * sizeof(float),
                stream>>>(A, A2, amode, segK, seg2K, rowOff, Wp(wt), Fp(bidx),
                          Fp(bidx + 1), Fp(bidx + 2), Y, N, K);
  };
  auto gemm = [&](const _Float16* A, const _Float16* A2, int amode, int segK,
                  int seg2K, int rowOff, int wt, int bidx, void* Y, int M,
                  int N, int K, int outmode) {
    k_gemm<<<dim3(M / 16, N / 16), dim3(32), 0, stream>>>(
        A, A2, amode, segK, seg2K, rowOff, Wp(wt), Fp(bidx), Y, N, K, outmode);
  };

  // ---- encoder: edge pipeline, 4 batch chunks of 64 (caps workspace) ----
  for (int cb = 0; cb < 4; ++cb) {
    int rowOff = cb * 15360;
    gemm_ln(nodes, nullptr, 1, 512, 0, rowOff, 0, 2, h1, 15360, 256, 1024);
    gemm_ln(h1, nullptr, 0, 256, 0, 0, 1, 6, h2, 15360, 256, 256);
    gemm(h2, nullptr, 0, 256, 0, 0, 2, 10, h1, 15360, 256, 256, 0);
    k_edge_mean<<<1024, 256, 0, stream>>>(h1, em, cb * 64);
  }
  // ---- node encoder ----
  gemm_ln(nodes, em, 2, 512, 256, 0, 3, 12, g1, 4096, 256, 768);
  gemm_ln(g1, nullptr, 0, 256, 0, 0, 4, 16, g2, 4096, 128, 256);
  gemm_ln(g2, nullptr, 0, 128, 0, 0, 5, 20, g1, 4096, 64, 128);
  gemm(g1, nullptr, 0, 64, 0, 0, 6, 24, zb, 4096, 32, 64, 0);
  // ---- decoder: edge pipeline ----
  for (int cb = 0; cb < 4; ++cb) {
    int rowOff = cb * 15360;
    gemm_ln(zb, nullptr, 1, 32, 0, rowOff, 7, 26, h1, 15360, 32, 64);
    gemm_ln(h1, nullptr, 0, 32, 0, 0, 8, 30, h2, 15360, 64, 32);
    gemm_ln(h2, nullptr, 0, 64, 0, 0, 9, 34, h1, 15360, 256, 64);
    gemm(h1, nullptr, 0, 256, 0, 0, 10, 38, h2, 15360, 256, 256, 0);
    k_edge_mean<<<1024, 256, 0, stream>>>(h2, emd, cb * 64);
  }
  // ---- node decoder (final store fuses transpose to (B,T,O,E) f32) ----
  gemm_ln(zb, emd, 2, 32, 256, 0, 11, 40, g1, 4096, 256, 288);
  gemm_ln(g1, nullptr, 0, 256, 0, 0, 12, 44, g2, 4096, 128, 256);
  gemm_ln(g2, nullptr, 0, 128, 0, 0, 13, 48, g1, 4096, 64, 128);
  gemm(g1, nullptr, 0, 64, 0, 0, 14, 52, d_out, 4096, 512, 64, 1);
}
